// Qwen35_80934363726551
// MI455X (gfx1250) — matrix operand order repaired and checked by reading, not compiled, not run on hardware
//
#include <hip/hip_runtime.h>
#include <hip/hip_bf16.h>

// Gated DeltaNet chunked scan for MI455X (gfx1250), wave32 + WMMA bf16.
// Grid: B*H workgroups (32). Block: 256 threads = 8 waves.
// Each wave owns a 16-wide Dv slice of the 128x128 recurrent state in VGPRs.
// All WMMA operands are laid out in LDS so fragment loads are contiguous
// b32/b64 DS ops; bf16 pair-packing via vector fptrunc -> v_cvt_pk_bf16_f32.

typedef __attribute__((ext_vector_type(16))) __bf16 v16bf;
typedef __attribute__((ext_vector_type(2)))  __bf16 v2bf;
typedef __attribute__((ext_vector_type(2)))  float  v2f;
typedef __attribute__((ext_vector_type(8)))  float  v8f;

#define Bn 2
#define Sn 4096
#define Hn 16
#define Dn 128
#define Cn 64
#define Nn (Sn / Cn)

#define LD   136
#define LDT  72
#define LD64 72
#define FLD  66

#define OFF_QN  0u
#define OFF_KN  17408u
#define OFF_KT  34816u
#define OFF_VT  53248u
#define OFF_KK  71680u
#define OFF_T   88576u
#define OFF_ATT 105472u
#define OFF_TB  114688u
#define OFF_TC  123904u
#define OFF_KC  133120u
#define OFF_ST  150528u
#define OFF_VN  185344u
#define OFF_GC  203776u
#define OFF_BT  204032u
#define OFF_QS  204288u
#define OFF_KS  204544u
#define OFF_CJ  204800u
#define SMEM_BYTES 205056u

static __device__ __forceinline__ unsigned pk2(float lo, float hi) {
#if __has_builtin(__builtin_amdgcn_cvt_pk_bf16_f32)
  auto t = __builtin_amdgcn_cvt_pk_bf16_f32(lo, hi);
  return __builtin_bit_cast(unsigned, t);
#else
  v2f f; f[0] = lo; f[1] = hi;
  v2bf t = __builtin_convertvector(f, v2bf);   // vector fptrunc -> v_cvt_pk_bf16_f32
  return __builtin_bit_cast(unsigned, t);
#endif
}
static __device__ __forceinline__ float lo16(unsigned u) {
  return __builtin_bit_cast(float, u << 16);
}
static __device__ __forceinline__ float hi16(unsigned u) {
  return __builtin_bit_cast(float, u & 0xFFFF0000u);
}

static __device__ __forceinline__ v16bf ld_a(const __bf16* base, int ld) {
  const int lane = threadIdx.x & 31;
  const int m = lane & 15, hf = lane >> 4;
  union { v16bf v; unsigned u[8]; } r;
#pragma unroll
  for (int vv = 0; vv < 8; ++vv) {
    const int kk = (vv << 1) + (hf << 4);
    r.u[vv] = *(const unsigned*)(base + m * ld + kk);
  }
  return r.v;
}
static __device__ __forceinline__ v16bf ld_bt(const __bf16* base, int ld) {
  const int lane = threadIdx.x & 31;
  const int n = lane & 15, hf = lane >> 4;
  const __bf16* p = base + n * ld + hf * 16;
  union { v16bf v; uint2 u[4]; } r;
#pragma unroll
  for (int j = 0; j < 4; ++j) r.u[j] = *(const uint2*)(p + 4 * j);
  return r.v;
}

static __device__ __forceinline__ v8f wmma_bf(v16bf a, v16bf b, v8f c) {
  return __builtin_amdgcn_wmma_f32_16x16x32_bf16(false, a, false, b, (short)0, c, false, false);
}

__global__ void __launch_bounds__(256)
gdn_chunk_scan(const float* __restrict__ q, const float* __restrict__ k,
               const float* __restrict__ v, const float* __restrict__ g,
               const float* __restrict__ beta, float* __restrict__ out) {
  extern __shared__ char smem[];
  __bf16* sQ  = (__bf16*)(smem + OFF_QN);
  __bf16* sK  = (__bf16*)(smem + OFF_KN);
  __bf16* sKT = (__bf16*)(smem + OFF_KT);
  __bf16* sVT = (__bf16*)(smem + OFF_VT);
  float*  fKK = (float*)(smem + OFF_KK);
  float*  fT  = (float*)(smem + OFF_T);
  __bf16* sAT = (__bf16*)(smem + OFF_ATT);
  __bf16* sTB = (__bf16*)(smem + OFF_TB);
  __bf16* sTC = (__bf16*)(smem + OFF_TC);
  __bf16* sKC = (__bf16*)(smem + OFF_KC);
  __bf16* sST = (__bf16*)(smem + OFF_ST);
  __bf16* sVN = (__bf16*)(smem + OFF_VN);
  float*  sGC = (float*)(smem + OFF_GC);
  float*  sBT = (float*)(smem + OFF_BT);
  float*  sQS = (float*)(smem + OFF_QS);
  float*  sKS = (float*)(smem + OFF_KS);
  float*  sCJ = (float*)(smem + OFF_CJ);

  const int bh = blockIdx.x;
  const int b  = bh / Hn, h = bh % Hn;
  const int tid  = threadIdx.x;
  const int wave = tid >> 5, lane = tid & 31;
  const int nn = lane & 15, h2 = lane >> 4;
  const float scale = 0.08838834764831845f;

  v8f st[8];
#pragma unroll
  for (int r = 0; r < 8; ++r) st[r] = v8f{};

  for (int nc = 0; nc < Nn; ++nc) {
    const int s0 = nc * Cn;

#pragma unroll
    for (int rr = 0; rr < 8; ++rr) {
      const int row = wave * 8 + rr;
      const size_t gbase = ((size_t)((b * Sn + s0 + row) * Hn + h)) * (size_t)Dn;
      float4 xq = ((const float4*)(q + gbase))[lane];
      float4 xk = ((const float4*)(k + gbase))[lane];
      float4 xv = ((const float4*)(v + gbase))[lane];
      float sq = xq.x * xq.x + xq.y * xq.y + xq.z * xq.z + xq.w * xq.w;
      float sk = xk.x * xk.x + xk.y * xk.y + xk.z * xk.z + xk.w * xk.w;
#pragma unroll
      for (int o = 16; o; o >>= 1) { sq += __shfl_xor(sq, o, 32); sk += __shfl_xor(sk, o, 32); }
      const float iq = 1.0f / fmaxf(sqrtf(sq), 1e-12f);
      const float ik = 1.0f / fmaxf(sqrtf(sk), 1e-12f);
      uint2 wq; wq.x = pk2(xq.x * iq, xq.y * iq); wq.y = pk2(xq.z * iq, xq.w * iq);
      uint2 wk; wk.x = pk2(xk.x * ik, xk.y * ik); wk.y = pk2(xk.z * ik, xk.w * ik);
      *(uint2*)(sQ + row * LD + lane * 4) = wq;
      *(uint2*)(sK + row * LD + lane * 4) = wk;
      const int d0 = lane * 4;
      sKT[(d0 + 0) * LDT + row] = (__bf16)(xk.x * ik);
      sKT[(d0 + 1) * LDT + row] = (__bf16)(xk.y * ik);
      sKT[(d0 + 2) * LDT + row] = (__bf16)(xk.z * ik);
      sKT[(d0 + 3) * LDT + row] = (__bf16)(xk.w * ik);
      sVT[(d0 + 0) * LDT + row] = (__bf16)xv.x;
      sVT[(d0 + 1) * LDT + row] = (__bf16)xv.y;
      sVT[(d0 + 2) * LDT + row] = (__bf16)xv.z;
      sVT[(d0 + 3) * LDT + row] = (__bf16)xv.w;
    }
    if (tid < Cn) {
      const size_t gi = (size_t)((b * Sn + s0 + tid) * Hn + h);
      sGC[tid] = g[gi];
      sBT[tid] = beta[gi];
    }
    __syncthreads();
    if (tid == 0) {
      float run = 0.f;
      for (int t = 0; t < Cn; ++t) { run += sGC[t]; sGC[t] = run; }
    }
    __syncthreads();

    if (tid < Cn) {
      const float gct = sGC[tid], glast = sGC[Cn - 1];
      sQS[tid] = scale * __expf(gct);
      sKS[tid] = __expf(glast - gct);
      sCJ[tid] = sBT[tid] * __expf(gct);
    }

#pragma unroll
    for (int qi = 0; qi < 4; ++qi) {
      const int id = wave * 4 + qi;
      const int isQK = id >> 4;
      const int tt = id & 15, ti = tt >> 2, tj = tt & 3;
      const __bf16* A = (isQK ? sQ : sK) + ti * 16 * LD;
      v8f c = v8f{};
#pragma unroll
      for (int kb = 0; kb < 4; ++kb)
        c = wmma_bf(ld_a(A + kb * 32, LD),
                    ld_bt(sK + tj * 16 * LD + kb * 32, LD), c);
      float* D = isQK ? fT : fKK;
#pragma unroll
      for (int e = 0; e < 8; ++e) D[(ti * 16 + e + 8 * h2) * FLD + tj * 16 + nn] = c[e];
    }
    __syncthreads();

    for (int idx = tid; idx < (Cn * Cn) / 2; idx += 256) {
      const int i = idx >> 5, jp = (idx & 31) << 1;
      const float gi = sGC[i], bi = sBT[i];
      float a0 = 0.f, a1 = 0.f, m0 = 0.f, m1 = 0.f;
      if (i >= jp) {
        const float d0 = __expf(gi - sGC[jp]);
        a0 = scale * fT[i * FLD + jp] * d0;
        m0 = (i > jp) ? -bi * fKK[i * FLD + jp] * d0 : 0.f;
      }
      if (i >= jp + 1) {
        const float d1 = __expf(gi - sGC[jp + 1]);
        a1 = scale * fT[i * FLD + jp + 1] * d1;
        m1 = (i > jp + 1) ? -bi * fKK[i * FLD + jp + 1] * d1 : 0.f;
      }
      *(unsigned*)(sAT + i * LD64 + jp) = pk2(a0, a1);
      float2 mm; mm.x = m0; mm.y = m1;
      *(float2*)(fKK + i * FLD + jp) = mm;
    }
    __syncthreads();

    if (tid < Cn) {
      const int j = tid;
      for (int i = 0; i < Cn; ++i) {
        float acc = (i == j) ? 1.f : 0.f;
        for (int l = j; l < i; ++l) acc += fKK[i * FLD + l] * fT[l * FLD + j];
        fT[i * FLD + j] = acc;
      }
    }
    __syncthreads();
    for (int idx = tid; idx < (Cn * Cn) / 2; idx += 256) {
      const int i = idx >> 5, jp = (idx & 31) << 1;
      const float t0 = fT[i * FLD + jp], t1 = fT[i * FLD + jp + 1];
      *(unsigned*)(sTB + i * LD64 + jp) = pk2(t0 * sBT[jp], t1 * sBT[jp + 1]);
      *(unsigned*)(sTC + i * LD64 + jp) = pk2(t0 * sCJ[jp], t1 * sCJ[jp + 1]);
    }
    __syncthreads();

#pragma unroll
    for (int qi = 0; qi < 4; ++qi) {
      const int id = wave * 4 + qi;
      const int ti = id >> 3, tj = id & 7;
      v8f c = v8f{};
#pragma unroll
      for (int kb = 0; kb < 2; ++kb)
        c = wmma_bf(ld_a(sTC + ti * 16 * LD64 + kb * 32, LD64),
                    ld_bt(sKT + tj * 16 * LDT + kb * 32, LDT), c);
#pragma unroll
      for (int e = 0; e < 8; ++e) sKC[(ti * 16 + e + 8 * h2) * LD + tj * 16 + nn] = (__bf16)c[e];
    }
    v8f vt[4];
#pragma unroll
    for (int ti = 0; ti < 4; ++ti) {
      v8f c = v8f{};
#pragma unroll
      for (int kb = 0; kb < 2; ++kb)
        c = wmma_bf(ld_a(sTB + ti * 16 * LD64 + kb * 32, LD64),
                    ld_bt(sVT + wave * 16 * LDT + kb * 32, LDT), c);
      vt[ti] = c;
    }
    __syncthreads();

    for (int idx = tid; idx < (Cn * Dn) / 2; idx += 256) {
      const int i = idx >> 6, dp = (idx & 63) << 1;
      unsigned* pq = (unsigned*)(sQ + i * LD + dp);
      const float qs = sQS[i];
      const unsigned uq = *pq;
      *pq = pk2(lo16(uq) * qs, hi16(uq) * qs);
      const int d = idx >> 5, tp = (idx & 31) << 1;
      unsigned* pk = (unsigned*)(sKT + d * LDT + tp);
      const unsigned uk = *pk;
      *pk = pk2(lo16(uk) * sKS[tp], hi16(uk) * sKS[tp + 1]);
    }
#pragma unroll
    for (int r = 0; r < 8; ++r) {
      uint4 w;
      w.x = pk2(st[r][0], st[r][1]); w.y = pk2(st[r][2], st[r][3]);
      w.z = pk2(st[r][4], st[r][5]); w.w = pk2(st[r][6], st[r][7]);
      *(uint4*)(sST + (wave * 16 + nn) * LD + r * 16 + 8 * h2) = w;
    }
    __syncthreads();

    v8f oacc[4];
#pragma unroll
    for (int ti = 0; ti < 4; ++ti) {
      v8f c = v8f{};
#pragma unroll
      for (int kb = 0; kb < 4; ++kb)
        c = wmma_bf(ld_a(sKC + ti * 16 * LD + kb * 32, LD),
                    ld_bt(sST + wave * 16 * LD + kb * 32, LD), c);
      float vn[8];
#pragma unroll
      for (int e = 0; e < 8; ++e) vn[e] = vt[ti][e] - c[e];
      uint4 w;
      w.x = pk2(vn[0], vn[1]); w.y = pk2(vn[2], vn[3]);
      w.z = pk2(vn[4], vn[5]); w.w = pk2(vn[6], vn[7]);
      *(uint4*)(sVN + (wave * 16 + nn) * LD64 + ti * 16 + 8 * h2) = w;
      v8f o = v8f{};
#pragma unroll
      for (int kb = 0; kb < 4; ++kb)
        o = wmma_bf(ld_a(sQ + ti * 16 * LD + kb * 32, LD),
                    ld_bt(sST + wave * 16 * LD + kb * 32, LD), o);
      oacc[ti] = o;
    }
#pragma unroll
    for (int ti = 0; ti < 4; ++ti) {
      v8f o = oacc[ti];
#pragma unroll
      for (int kb = 0; kb < 2; ++kb)
        o = wmma_bf(ld_a(sAT + ti * 16 * LD64 + kb * 32, LD64),
                    ld_bt(sVN + wave * 16 * LD64 + kb * 32, LD64), o);
#pragma unroll
      for (int e = 0; e < 8; ++e) {
        const int row = ti * 16 + e + 8 * h2;
        const size_t go = ((size_t)((b * Sn + s0 + row) * Hn + h)) * (size_t)Dn + wave * 16 + nn;
        out[go] = o[e];
      }
    }
    const float dsc = __expf(sGC[Cn - 1]);
#pragma unroll
    for (int r = 0; r < 8; ++r) {
      v8f c = st[r];
#pragma unroll
      for (int e = 0; e < 8; ++e) c[e] *= dsc;
#pragma unroll
      for (int kb = 0; kb < 2; ++kb)
        c = wmma_bf(ld_a(sKT + r * 16 * LDT + kb * 32, LDT),
                    ld_bt(sVN + wave * 16 * LD64 + kb * 32, LD64), c);
      st[r] = c;
    }
    __syncthreads();
  }
}

extern "C" void kernel_launch(void* const* d_in, const int* in_sizes, int n_in,
                              void* d_out, int out_size, void* d_ws, size_t ws_size,
                              hipStream_t stream) {
  (void)in_sizes; (void)n_in; (void)out_size; (void)d_ws; (void)ws_size;
  const float* q    = (const float*)d_in[0];
  const float* k    = (const float*)d_in[1];
  const float* v    = (const float*)d_in[2];
  const float* g    = (const float*)d_in[3];
  const float* beta = (const float*)d_in[4];
  float* out = (float*)d_out;
  dim3 grid(Bn * Hn), block(256);
  hipLaunchKernelGGL(gdn_chunk_scan, grid, block, SMEM_BYTES, stream, q, k, v, g, beta, out);
}
